// OurModel_43903155699746
// MI455X (gfx1250) — matrix operand order repaired and checked by reading, not compiled, not run on hardware
//
#include <hip/hip_runtime.h>
#include <cmath>

// ---------------------------------------------------------------------------
// BiRNN (tanh) + tag projection + softmax + CRF log-likelihood for MI455X.
// All three GEMMs run on v_wmma_f32_16x16x32_f16 (f16 in, f32 accumulate).
// ---------------------------------------------------------------------------

typedef _Float16 v16h __attribute__((ext_vector_type(16)));
typedef _Float16 v8h  __attribute__((ext_vector_type(8)));
typedef float    v8f  __attribute__((ext_vector_type(8)));

#define S_LEN 512
#define B_LEN 256
#define D_IN  202
#define H_DIM 100
#define T_TAG 19

__device__ __forceinline__ v8f wmma_f16(v16h a, v16h b, v8f c) {
    // D = A(16x32,f16) x B(32x16,f16) + C(16x16,f32)
    return __builtin_amdgcn_wmma_f32_16x16x32_f16(
        /*neg_a=*/false, a, /*neg_b=*/false, b,
        /*c_mod=*/(short)0, c, /*reuse_a=*/false, /*reuse_b=*/false);
}

// A-fragment (16x32 f16, row-major source with row stride `ld` halves).
// Lane layout (ISA 7.12.2): m = lane&15, hi = lane>>4;
// elems 0..7  <- K = kt*32 + hi*8 + {0..7}
// elems 8..15 <- K = kt*32 + 16 + hi*8 + {0..7}
__device__ __forceinline__ v16h load_a_frag(const _Float16* __restrict__ base,
                                            int ld, int m0, int kt, int lane) {
    int m  = m0 + (lane & 15);
    int hi = lane >> 4;
    const _Float16* p = base + (size_t)m * ld + kt * 32 + hi * 8;
    v8h c0 = *(const v8h*)(p);
    v8h c1 = *(const v8h*)(p + 16);
    return __builtin_shufflevector(c0, c1, 0,1,2,3,4,5,6,7,8,9,10,11,12,13,14,15);
}

// Same, but substitutes zero chunks when K-range exceeds kmax (row width).
__device__ __forceinline__ v16h load_a_frag_ck(const _Float16* __restrict__ base,
                                               int ld, int m0, int kt, int lane,
                                               int kmax) {
    int m  = m0 + (lane & 15);
    int hi = lane >> 4;
    int k0 = kt * 32 + hi * 8;
    const _Float16* p = base + (size_t)m * ld + k0;
    v8h z = {};
    v8h c0 = z, c1 = z;
    if (k0 < kmax)      c0 = *(const v8h*)(p);
    if (k0 + 16 < kmax) c1 = *(const v8h*)(p + 16);
    return __builtin_shufflevector(c0, c1, 0,1,2,3,4,5,6,7,8,9,10,11,12,13,14,15);
}

// ---------------------------------------------------------------------------
// Pack a row-major fp32 weight W[N][K] into fragment-major f16 B-tiles.
// Tile (kt,nt) -> 512 halves; per lane: n = nt*16 + (lane&15),
// elem e<8 -> k = kt*32 + (lane>>4)*8 + e; e>=8 -> k = kt*32 + 16 + (lane>>4)*8 + (e-8).  Zero padded beyond K,N.
// ---------------------------------------------------------------------------
__global__ void pack_b_kernel(const float* __restrict__ src, _Float16* __restrict__ dst,
                              int K, int N, int ktiles, int ntiles) {
    int gid = blockIdx.x * 256 + threadIdx.x;
    int total = ktiles * ntiles * 512;
    if (gid >= total) return;
    int tile = gid >> 9;
    int r    = gid & 511;
    int lane = r >> 4, e = r & 15;
    int kt = tile / ntiles, nt = tile - kt * ntiles;
    int n  = nt * 16 + (lane & 15);
    int k  = kt * 32 + (lane >> 4) * 8 + (e & 7) + (e >> 3) * 16;
    float v = (k < K && n < N) ? src[(size_t)n * K + k] : 0.0f;
    dst[gid] = (_Float16)v;
}

// ---------------------------------------------------------------------------
// Input projection, both directions fused:
// xp[dir][s*B+b][n] = f16( sum_d x[s,b,d]*Wih[n,d] + bih[n] + bhh[n] )
// Grid: 1024 blocks x 256 threads; each wave owns one 16-row M-tile.
// ---------------------------------------------------------------------------
__global__ __launch_bounds__(256) void xproj_kernel(
    const float* __restrict__ x,
    const _Float16* __restrict__ wihF, const _Float16* __restrict__ wihB,
    const float* __restrict__ bihF, const float* __restrict__ bhhF,
    const float* __restrict__ bihB, const float* __restrict__ bhhB,
    _Float16* __restrict__ xpF, _Float16* __restrict__ xpB) {
    __shared__ _Float16 xa[8][16 * 224];
    int tid = threadIdx.x, wave = tid >> 5, lane = tid & 31;
    int mt = blockIdx.x * 8 + wave;   // 0..8191
    int M0 = mt * 16;

    _Float16* lds = xa[wave];
    for (int idx = lane; idx < 16 * 224; idx += 32) {
        int r = idx / 224, c = idx - r * 224;
        float v = (c < D_IN) ? x[(size_t)(M0 + r) * D_IN + c] : 0.0f;
        lds[r * 224 + c] = (_Float16)v;
    }
    // per-wave LDS: in-order DS pipe, no workgroup barrier needed.

    int hi = lane >> 4, nlo = lane & 15;
    for (int dir = 0; dir < 2; ++dir) {
        const _Float16* wih = dir ? wihB : wihF;
        const float* bih = dir ? bihB : bihF;
        const float* bhh = dir ? bhhB : bhhF;
        _Float16* xp = dir ? xpB : xpF;
        v8f acc[7];
#pragma unroll
        for (int nt = 0; nt < 7; ++nt) acc[nt] = (v8f){};
#pragma unroll
        for (int kt = 0; kt < 7; ++kt) {
            v16h a = load_a_frag(lds, 224, 0, kt, lane);
#pragma unroll
            for (int nt = 0; nt < 7; ++nt) {
                v16h b = *(const v16h*)(wih + (kt * 7 + nt) * 512 + lane * 16);
                acc[nt] = wmma_f16(a, b, acc[nt]);
            }
        }
#pragma unroll
        for (int nt = 0; nt < 7; ++nt) {
            int n = nt * 16 + nlo;
            if (n < H_DIM) {
#pragma unroll
                for (int r = 0; r < 8; ++r) {
                    int m = r + hi * 8;
                    float v = acc[nt][r] + bih[n] + bhh[n];
                    xp[(size_t)(M0 + m) * H_DIM + n] = (_Float16)v;
                }
            }
        }
    }
}

// ---------------------------------------------------------------------------
// Recurrent scan: h_t = tanh(xp_t + h_{t-1} @ Whh^T). One persistent block
// per direction; h kept in LDS as f16 [256][112] (zero-padded columns).
// 28 waves x 4 tiles x 4 k-steps = 448 WMMAs per step.
// ---------------------------------------------------------------------------
__global__ __launch_bounds__(896) void rnn_scan_kernel(
    const _Float16* __restrict__ xpF, const _Float16* __restrict__ xpB,
    const _Float16* __restrict__ whhF, const _Float16* __restrict__ whhB,
    _Float16* __restrict__ hout /* [S][B][200] */) {
    __shared__ _Float16 hbuf[256 * 112];
    int tid = threadIdx.x, wave = tid >> 5, lane = tid & 31;
    int dir = blockIdx.x;
    const _Float16* xp  = dir ? xpB  : xpF;
    const _Float16* whh = dir ? whhB : whhF;

    for (int i = tid; i < 256 * 112; i += 896) hbuf[i] = (_Float16)0.f;
    __syncthreads();

    int hi = lane >> 4, nlo = lane & 15;
    for (int step = 0; step < S_LEN; ++step) {
        int s = dir ? (S_LEN - 1 - step) : step;
        v8f acc[4];
        int mts[4], nts[4];
#pragma unroll
        for (int i = 0; i < 4; ++i) {
            int t  = wave + 28 * i;          // 0..111
            int mt = t / 7, nt = t - mt * 7;
            mts[i] = mt; nts[i] = nt;
            v8f c = {};
#pragma unroll
            for (int kt = 0; kt < 4; ++kt) {
                v16h a = load_a_frag_ck(hbuf, 112, mt * 16, kt, lane, 112);
                v16h b = *(const v16h*)(whh + (kt * 7 + nt) * 512 + lane * 16);
                c = wmma_f16(a, b, c);
            }
            acc[i] = c;
        }
        __syncthreads();   // all reads of h_{t-1} done before overwriting
#pragma unroll
        for (int i = 0; i < 4; ++i) {
            int mt = mts[i], nt = nts[i];
            int n  = nt * 16 + nlo;
            if (n < H_DIM) {
#pragma unroll
                for (int r = 0; r < 8; ++r) {
                    int m = r + hi * 8;
                    int b = mt * 16 + m;
                    float xv = (float)xp[((size_t)s * B_LEN + b) * H_DIM + n];
                    float hv = tanhf(acc[i][r] + xv);
                    _Float16 h16 = (_Float16)hv;
                    hbuf[b * 112 + n] = h16;
                    hout[((size_t)s * B_LEN + b) * (2 * H_DIM) + dir * H_DIM + n] = h16;
                }
            }
        }
        __syncthreads();
    }
}

// ---------------------------------------------------------------------------
// Per-sequence (512 blocks): tag projection via WMMA, softmax, CRF llh.
// ---------------------------------------------------------------------------
__global__ __launch_bounds__(256) void crf_kernel(
    const _Float16* __restrict__ hglob, const _Float16* __restrict__ wtag,
    const float* __restrict__ btag, const int* __restrict__ y,
    const float* __restrict__ start_t, const float* __restrict__ end_t,
    const float* __restrict__ trans, float* __restrict__ llh) {
    __shared__ float pr[256][20];
    __shared__ float tr[T_TAG * T_TAG];
    __shared__ float stv[T_TAG], env[T_TAG], btv[T_TAG];
    __shared__ int   ytag[256];
    __shared__ float red[256];
    __shared__ float al[2][T_TAG];
    int tid = threadIdx.x, wave = tid >> 5, lane = tid & 31;
    int s = blockIdx.x;

    if (tid < T_TAG) { stv[tid] = start_t[tid]; env[tid] = end_t[tid]; btv[tid] = btag[tid]; }
    for (int i = tid; i < T_TAG * T_TAG; i += 256) tr[i] = trans[i];
    ytag[tid] = y[s * B_LEN + tid];
    __syncthreads();

    // logits = h[s] @ Wtag^T + btag  -> pr (still logits here)
    const _Float16* hbase = hglob + (size_t)s * B_LEN * (2 * H_DIM);
#pragma unroll
    for (int i = 0; i < 4; ++i) {
        int t  = wave * 4 + i;   // 0..31
        int mt = t >> 1, nt = t & 1;
        v8f c = {};
#pragma unroll
        for (int kt = 0; kt < 7; ++kt) {
            v16h a = load_a_frag(hbase, 2 * H_DIM, mt * 16, kt, lane);
            v16h b = *(const v16h*)(wtag + (kt * 2 + nt) * 512 + lane * 16);
            c = wmma_f16(a, b, c);
        }
        int n = nt * 16 + (lane & 15);
        if (n < T_TAG) {
#pragma unroll
            for (int r = 0; r < 8; ++r) {
                int m = r + (lane >> 4) * 8;
                pr[mt * 16 + m][n] = c[r] + btv[n];
            }
        }
    }
    __syncthreads();

    // softmax over 19 tags; row b = tid
    {
        float mx = -1e30f;
        for (int t2 = 0; t2 < T_TAG; ++t2) mx = fmaxf(mx, pr[tid][t2]);
        float e[T_TAG], den = 0.f;
        for (int t2 = 0; t2 < T_TAG; ++t2) { e[t2] = expf(pr[tid][t2] - mx); den += e[t2]; }
        float inv = 1.0f / den;
        for (int t2 = 0; t2 < T_TAG; ++t2) pr[tid][t2] = e[t2] * inv;
    }
    __syncthreads();

    // numerator (gold path score), deterministic tree reduction
    {
        int yb = ytag[tid];
        float part = pr[tid][yb];
        if (tid == 0) part += stv[yb];
        else          part += tr[ytag[tid - 1] * T_TAG + yb];
        if (tid == B_LEN - 1) part += env[yb];
        red[tid] = part;
    }
    for (int off = 128; off > 0; off >>= 1) {
        __syncthreads();
        if (tid < off) red[tid] += red[tid + off];
    }
    __syncthreads();

    // denominator: forward algorithm, lanes 0..18 of wave 0 (lockstep wave32)
    if (tid < T_TAG) {
        al[0][tid] = stv[tid] + pr[0][tid];
        int cur = 0;
        for (int b = 1; b < B_LEN; ++b) {
            float mx = -1e30f;
            for (int i2 = 0; i2 < T_TAG; ++i2)
                mx = fmaxf(mx, al[cur][i2] + tr[i2 * T_TAG + tid]);
            float sum = 0.f;
            for (int i2 = 0; i2 < T_TAG; ++i2)
                sum += expf(al[cur][i2] + tr[i2 * T_TAG + tid] - mx);
            al[1 - cur][tid] = mx + logf(sum) + pr[b][tid];
            cur ^= 1;
        }
        if (tid == 0) {
            float mx = -1e30f;
            for (int j = 0; j < T_TAG; ++j) mx = fmaxf(mx, al[cur][j] + env[j]);
            float sum = 0.f;
            for (int j = 0; j < T_TAG; ++j) sum += expf(al[cur][j] + env[j] - mx);
            llh[s] = red[0] - (mx + logf(sum));
        }
    }
}

// Deterministic final sum over 512 sequences.
__global__ void sum_kernel(const float* __restrict__ llh, float* __restrict__ out) {
    if (threadIdx.x == 0) {
        float s = 0.f;
        for (int i = 0; i < S_LEN; ++i) s += llh[i];
        out[0] = s;
    }
}

// ---------------------------------------------------------------------------
extern "C" void kernel_launch(void* const* d_in, const int* in_sizes, int n_in,
                              void* d_out, int out_size, void* d_ws, size_t ws_size,
                              hipStream_t stream) {
    (void)in_sizes; (void)n_in; (void)out_size; (void)ws_size;
    const float* x      = (const float*)d_in[0];
    const int*   y      = (const int*)  d_in[1];
    const float* Wih_f  = (const float*)d_in[2];
    const float* Whh_f  = (const float*)d_in[3];
    const float* bih_f  = (const float*)d_in[4];
    const float* bhh_f  = (const float*)d_in[5];
    const float* Wih_b  = (const float*)d_in[6];
    const float* Whh_b  = (const float*)d_in[7];
    const float* bih_b  = (const float*)d_in[8];
    const float* bhh_b  = (const float*)d_in[9];
    const float* W_tag  = (const float*)d_in[10];
    const float* b_tag  = (const float*)d_in[11];
    const float* startt = (const float*)d_in[12];
    const float* endt   = (const float*)d_in[13];
    const float* transm = (const float*)d_in[14];

    // workspace layout (256B aligned)
    char* ws = (char*)d_ws;
    size_t off = 0;
    auto alloc = [&](size_t bytes) { size_t o = off; off += (bytes + 255) & ~(size_t)255; return o; };
    const size_t XP_B   = (size_t)S_LEN * B_LEN * H_DIM * 2;       // 25 MiB each
    const size_t H_B    = (size_t)S_LEN * B_LEN * 2 * H_DIM * 2;   // 50 MiB
    _Float16* xpF  = (_Float16*)(ws + alloc(XP_B));
    _Float16* xpB  = (_Float16*)(ws + alloc(XP_B));
    _Float16* hbuf = (_Float16*)(ws + alloc(H_B + 256));           // +pad for frag overreads
    _Float16* wihF = (_Float16*)(ws + alloc(7 * 7 * 512 * 2));
    _Float16* wihB = (_Float16*)(ws + alloc(7 * 7 * 512 * 2));
    _Float16* whhF = (_Float16*)(ws + alloc(4 * 7 * 512 * 2));
    _Float16* whhB = (_Float16*)(ws + alloc(4 * 7 * 512 * 2));
    _Float16* wtag = (_Float16*)(ws + alloc(7 * 2 * 512 * 2));
    float*    llh  = (float*)   (ws + alloc(S_LEN * 4));

    // pack weights into WMMA B-fragment layout
    pack_b_kernel<<<(7 * 7 * 512 + 255) / 256, 256, 0, stream>>>(Wih_f, wihF, D_IN, H_DIM, 7, 7);
    pack_b_kernel<<<(7 * 7 * 512 + 255) / 256, 256, 0, stream>>>(Wih_b, wihB, D_IN, H_DIM, 7, 7);
    pack_b_kernel<<<(4 * 7 * 512 + 255) / 256, 256, 0, stream>>>(Whh_f, whhF, H_DIM, H_DIM, 4, 7);
    pack_b_kernel<<<(4 * 7 * 512 + 255) / 256, 256, 0, stream>>>(Whh_b, whhB, H_DIM, H_DIM, 4, 7);
    pack_b_kernel<<<(7 * 2 * 512 + 255) / 256, 256, 0, stream>>>(W_tag, wtag, 2 * H_DIM, T_TAG, 7, 2);

    // input projection (both directions), WMMA GEMM
    xproj_kernel<<<dim3(1024), 256, 0, stream>>>(x, wihF, wihB, bih_f, bhh_f, bih_b, bhh_b, xpF, xpB);

    // recurrent scans: one persistent workgroup per direction
    rnn_scan_kernel<<<2, 896, 0, stream>>>(xpF, xpB, whhF, whhB, hbuf);

    // tag projection + softmax + CRF per sequence
    crf_kernel<<<S_LEN, 256, 0, stream>>>(hbuf, wtag, b_tag, y, startt, endt, transm, llh);

    // deterministic scalar reduction
    sum_kernel<<<1, 32, 0, stream>>>(llh, (float*)d_out);
}
